// Transformer_80917183856745
// MI455X (gfx1250) — matrix operand order repaired and checked by reading, not compiled, not run on hardware
//
#include <hip/hip_runtime.h>
#include <hip/hip_bf16.h>
#include <math.h>

#define DEPTH 6
#define HEADS 16
#define DIM_HEAD 64
#define DIM 1024
#define INNER (HEADS * DIM_HEAD)   // 1024
#define MLP DIM
#define SEQ 1024
#define BATCH 8
#define MTOT (BATCH * SEQ)         // 8192
#define ATT_SCALE 0.125f           // 64^-0.5
#define LN_EPS 1e-5f

typedef __attribute__((ext_vector_type(16))) _Float16 v16h;
typedef __attribute__((ext_vector_type(8)))  _Float16 h8;
typedef __attribute__((ext_vector_type(4)))  _Float16 h4;
typedef __attribute__((ext_vector_type(8)))  float    v8f;

__device__ __forceinline__ v8f wmma_f16f32(v16h a, v16h b, v8f c) {
  // D = A(16x32 f16) * B(32x16 f16) + C(16x16 f32)
  return __builtin_amdgcn_wmma_f32_16x16x32_f16(false, a, false, b, (short)0, c,
                                                false, false);
}

// CDNA5 async memory->LDS copy (16B per lane), tracked by ASYNCcnt.
// vdst = per-lane LDS byte address (low 32 bits of the generic pointer),
// vaddr = per-lane 64-bit global address.
__device__ __forceinline__ void async_load_b128(const void* gptr, void* lptr) {
  const unsigned lds = (unsigned)(__SIZE_TYPE__)lptr;
  const unsigned long long gaddr = (unsigned long long)(__SIZE_TYPE__)gptr;
  asm volatile("global_load_async_to_lds_b128 %0, %1, off"
               :: "v"(lds), "v"(gaddr) : "memory");
}
__device__ __forceinline__ void wait_asynccnt0() {
  asm volatile("s_wait_asynccnt 0x0" ::: "memory");
}

// A fragment (16x32, row-major source [row][k], row stride `ld` halves).
// ISA layout: lanes 0-15 hold row=lane, K = 0..7 (v0..3) and 16..23 (v4..7);
// lanes 16-31 hold the same rows, K = 8..15 and 24..31.
__device__ __forceinline__ v16h load_a_frag(const _Float16* base, int ld) {
  const int lane = threadIdx.x & 31;
  const int row  = lane & 15;
  const int kh   = (lane & 16) ? 8 : 0;
  const _Float16* p = base + (size_t)row * ld + kh;
  h8 lo = *(const h8*)(p);
  h8 hi = *(const h8*)(p + 16);
  v16h r;
#pragma unroll
  for (int i = 0; i < 8; ++i) { r[i] = lo[i]; r[i + 8] = hi[i]; }
  return r;
}

// B fragment (32x16). Source stored "B-transposed": [n][k] row-major with row
// stride `ld`. Layout: lanes 0-15 hold col=lane, K=0..7 and 16..23; lanes 16-31
// hold col=lane-16, K=8..15 and 24..31.
__device__ __forceinline__ v16h load_b_frag(const _Float16* base, int ld) {
  const int lane = threadIdx.x & 31;
  const int col  = lane & 15;
  const int kh   = (lane & 16) ? 8 : 0;
  const _Float16* p = base + (size_t)col * ld + kh;
  h8 lo = *(const h8*)(p);
  h8 hi = *(const h8*)(p + 16);
  v16h r;
#pragma unroll
  for (int i = 0; i < 8; ++i) { r[i] = lo[i]; r[i + 8] = hi[i]; }
  return r;
}

// ---------------------------------------------------------------------------
// RPQ weight dequantization: W[o, g*64+j] = cb[g, codes[o,g], j]  -> f16
// rows 0..3071 -> Wqkv, 3072..4095 -> Wout, 4096..5119 -> Wfc1, 5120..6143 -> Wfc2
// ---------------------------------------------------------------------------
__global__ __launch_bounds__(256) void build_weights(
    const float* __restrict__ cb0, const float* __restrict__ cb1,
    const float* __restrict__ cb2, const float* __restrict__ cb3,
    const int* __restrict__ cq, const int* __restrict__ co,
    const int* __restrict__ c1, const int* __restrict__ c2,
    _Float16* __restrict__ wqkv, _Float16* __restrict__ wout,
    _Float16* __restrict__ wfc1, _Float16* __restrict__ wfc2) {
  const int row = blockIdx.x;   // 0..6143
  const int tid = threadIdx.x;  // 256 threads, 4 elems each
  const int e0  = tid * 4;
  const int g   = e0 >> 6;
  const int j0  = e0 & 63;

  const float* cb; const int* codes; _Float16* dst; int r;
  if (row < 3 * INNER)                 { cb = cb0; codes = cq; dst = wqkv; r = row; }
  else if (row < 3 * INNER + DIM)      { cb = cb1; codes = co; dst = wout; r = row - 3 * INNER; }
  else if (row < 3 * INNER + DIM + MLP){ cb = cb2; codes = c1; dst = wfc1; r = row - 3 * INNER - DIM; }
  else                                 { cb = cb3; codes = c2; dst = wfc2; r = row - 3 * INNER - DIM - MLP; }

  const int code = codes[(size_t)r * HEADS + g];
  const float4 v = *(const float4*)(cb + ((size_t)g * 256 + code) * 64 + j0);
  h4 o;
  o[0] = (_Float16)v.x; o[1] = (_Float16)v.y;
  o[2] = (_Float16)v.z; o[3] = (_Float16)v.w;
  *(h4*)(dst + (size_t)r * DIM + e0) = o;
}

// ---------------------------------------------------------------------------
// LayerNorm (fp32 in) -> f16 out, one row (1024) per block
// ---------------------------------------------------------------------------
__global__ __launch_bounds__(256) void layernorm_to_f16(
    const float* __restrict__ x, const float* __restrict__ w,
    const float* __restrict__ b, _Float16* __restrict__ out) {
  const int row = blockIdx.x;
  const int tid = threadIdx.x;
  const float4 v = *(const float4*)(x + (size_t)row * DIM + tid * 4);
  float s  = v.x + v.y + v.z + v.w;
  float s2 = v.x * v.x + v.y * v.y + v.z * v.z + v.w * v.w;
#pragma unroll
  for (int off = 16; off > 0; off >>= 1) {
    s  += __shfl_xor(s,  off, 32);
    s2 += __shfl_xor(s2, off, 32);
  }
  __shared__ float red[2][8];
  if ((tid & 31) == 0) { red[0][tid >> 5] = s; red[1][tid >> 5] = s2; }
  __syncthreads();
  float ts = 0.f, ts2 = 0.f;
#pragma unroll
  for (int j = 0; j < 8; ++j) { ts += red[0][j]; ts2 += red[1][j]; }
  const float mean = ts * (1.0f / DIM);
  const float var  = ts2 * (1.0f / DIM) - mean * mean;
  const float rstd = rsqrtf(var + LN_EPS);
  const float* wr = w + tid * 4;
  const float* br = b + tid * 4;
  const float xv[4] = {v.x, v.y, v.z, v.w};
  h4 o;
#pragma unroll
  for (int j = 0; j < 4; ++j)
    o[j] = (_Float16)((xv[j] - mean) * rstd * wr[j] + br[j]);
  *(h4*)(out + (size_t)row * DIM + tid * 4) = o;
}

// ---------------------------------------------------------------------------
// Generic WMMA GEMM:  out[M,N] = A[M,K](f16) * W[N,K]^T(f16)  (+bias)(+gelu)(+res)
// 128x128x32 block tile, 8 waves, each wave 32x64 (2x4 WMMA tiles).
// Double-buffered LDS staging via global_load_async_to_lds_b128 (ASYNCcnt):
// tile k+1 is in flight while tile k's WMMAs execute; one barrier per K-step.
// ---------------------------------------------------------------------------
#define GBM 128
#define GBN 128
#define GBK 32
#define GF_BIAS 1
#define GF_GELU 2
#define GF_RES  4
#define GF_F32  8

__global__ __launch_bounds__(256) void gemm_wmma(
    const _Float16* __restrict__ A, const _Float16* __restrict__ W,
    const float* __restrict__ bias, const float* __restrict__ residual,
    float* __restrict__ outF, _Float16* __restrict__ outH,
    int M, int N, int K, int flags) {
  __shared__ _Float16 As[2][GBM][GBK];
  __shared__ _Float16 Bs[2][GBN][GBK];
  const int tid  = threadIdx.x;
  const int lane = tid & 31;
  const int wid  = tid >> 5;
  const int wm   = wid >> 1;  // 0..3 : 32-row strip
  const int wn   = wid & 1;   // 0..1 : 64-col strip
  const int m0   = blockIdx.y * GBM;
  const int n0   = blockIdx.x * GBN;

  // Each thread stages 2x 16B for A and 2x 16B for B per tile.
  const int srow0 = tid >> 2;              // 0..63
  const int srow1 = srow0 + 64;            // 64..127
  const int scg   = (tid & 3) * 8;         // 0,8,16,24

  v8f acc[2][4];
#pragma unroll
  for (int i = 0; i < 2; ++i)
#pragma unroll
    for (int j = 0; j < 4; ++j) acc[i][j] = {};

  // Prologue: stage tile 0 into buffer 0.
  async_load_b128(A + (size_t)(m0 + srow0) * K + scg, &As[0][srow0][scg]);
  async_load_b128(A + (size_t)(m0 + srow1) * K + scg, &As[0][srow1][scg]);
  async_load_b128(W + (size_t)(n0 + srow0) * K + scg, &Bs[0][srow0][scg]);
  async_load_b128(W + (size_t)(n0 + srow1) * K + scg, &Bs[0][srow1][scg]);
  wait_asynccnt0();
  __syncthreads();

  const int nk = K / GBK;
  for (int kt = 0; kt < nk; ++kt) {
    const int cur = kt & 1;
    if (kt + 1 < nk) {  // stage next tile into the other buffer (async)
      const int nxt = cur ^ 1;
      const int k1  = (kt + 1) * GBK;
      async_load_b128(A + (size_t)(m0 + srow0) * K + k1 + scg, &As[nxt][srow0][scg]);
      async_load_b128(A + (size_t)(m0 + srow1) * K + k1 + scg, &As[nxt][srow1][scg]);
      async_load_b128(W + (size_t)(n0 + srow0) * K + k1 + scg, &Bs[nxt][srow0][scg]);
      async_load_b128(W + (size_t)(n0 + srow1) * K + k1 + scg, &Bs[nxt][srow1][scg]);
    }
    v16h afrag[2];
#pragma unroll
    for (int tm = 0; tm < 2; ++tm)
      afrag[tm] = load_a_frag(&As[cur][wm * 32 + tm * 16][0], GBK);
#pragma unroll
    for (int tn = 0; tn < 4; ++tn) {
      const v16h bfrag = load_b_frag(&Bs[cur][wn * 64 + tn * 16][0], GBK);
#pragma unroll
      for (int tm = 0; tm < 2; ++tm)
        acc[tm][tn] = wmma_f16f32(afrag[tm], bfrag, acc[tm][tn]);
    }
    wait_asynccnt0();   // next tile landed in LDS
    __syncthreads();    // and everyone is done reading the current buffer
  }

  // Epilogue (C layout: VGPR r -> row r / r+8; lane&15 -> col)
  const int colg  = lane & 15;
  const int rhalf = (lane & 16) ? 8 : 0;
#pragma unroll
  for (int tm = 0; tm < 2; ++tm) {
#pragma unroll
    for (int tn = 0; tn < 4; ++tn) {
#pragma unroll
      for (int r = 0; r < 8; ++r) {
        const int m = m0 + wm * 32 + tm * 16 + rhalf + r;
        const int n = n0 + wn * 64 + tn * 16 + colg;
        float v = acc[tm][tn][r];
        if (flags & GF_BIAS) v += bias[n];
        if (flags & GF_GELU) v = 0.5f * v * (1.0f + erff(v * 0.70710678f));
        if (flags & GF_RES)  v += residual[(size_t)m * N + n];
        if (flags & GF_F32)  outF[(size_t)m * N + n] = v;
        else                 outH[(size_t)m * N + n] = (_Float16)v;
      }
    }
  }
}

// ---------------------------------------------------------------------------
// Fused flash attention. Block = (qblock of 128, head, batch); 8 waves,
// each wave owns a 16-row q tile. K/V streamed in 64-key chunks through LDS.
// K tile staged with async-to-LDS; V transposed through ds stores.
// qkv layout: (b, n, 3*INNER) f16 with q|k|v concatenated per row.
// ---------------------------------------------------------------------------
__global__ __launch_bounds__(256) void attention_kernel(
    const _Float16* __restrict__ qkv, _Float16* __restrict__ attnout) {
  __shared__ _Float16 Kt[64][64];      // [key][d]
  __shared__ _Float16 Vt[64][72];      // [d][key]  (transposed, padded)
  __shared__ _Float16 Pb[8][16][72];   // per-wave P staging (C->A relayout)

  const int tid   = threadIdx.x;
  const int lane  = tid & 31;
  const int wid   = tid >> 5;
  const int qb    = blockIdx.x;        // 0..7
  const int h     = blockIdx.y;        // 0..15
  const int b     = blockIdx.z;        // 0..7
  const int colg  = lane & 15;
  const int rhalf = (lane & 16) ? 8 : 0;

  const size_t rs = 3 * INNER;  // qkv row stride (halves)
  const _Float16* qbase =
      qkv + ((size_t)b * SEQ + qb * 128 + wid * 16) * rs + h * DIM_HEAD;
  const v16h Qa0 = load_a_frag(qbase, (int)rs);        // d = 0..31
  const v16h Qa1 = load_a_frag(qbase + 32, (int)rs);   // d = 32..63

  float mrow[8], lrow[8];
#pragma unroll
  for (int r = 0; r < 8; ++r) { mrow[r] = -1e30f; lrow[r] = 0.f; }
  v8f Oacc[4];
#pragma unroll
  for (int t = 0; t < 4; ++t) Oacc[t] = {};

  for (int kc = 0; kc < SEQ / 64; ++kc) {
    __syncthreads();  // protect Kt/Vt against previous iteration's readers
#pragma unroll
    for (int it = 0; it < 2; ++it) {
      const int s  = tid + it * 256;   // 512 tasks = 64 keys x 8 groups of 8
      const int kr = s >> 3;
      const int cg = (s & 7) * 8;
      const _Float16* kp =
          qkv + ((size_t)b * SEQ + kc * 64 + kr) * rs + INNER + h * DIM_HEAD + cg;
      async_load_b128(kp, &Kt[kr][cg]);            // K tile: memory -> LDS direct
      const h8 vv = *(const h8*)(kp + INNER);      // V tile: transpose via VGPRs
#pragma unroll
      for (int j = 0; j < 8; ++j) Vt[cg + j][kr] = vv[j];
    }
    wait_asynccnt0();
    __syncthreads();

    // S(16x64) = Q * K^T
    v8f sc[4];
#pragma unroll
    for (int t = 0; t < 4; ++t) {
      v8f z = {};
      z = wmma_f16f32(Qa0, load_b_frag(&Kt[t * 16][0], 64), z);
      z = wmma_f16f32(Qa1, load_b_frag(&Kt[t * 16][32], 64), z);
      sc[t] = z;
    }

    // online softmax (reduce across the 16-lane column groups)
#pragma unroll
    for (int r = 0; r < 8; ++r) {
      float mx = -1e30f;
#pragma unroll
      for (int t = 0; t < 4; ++t) { sc[t][r] *= ATT_SCALE; mx = fmaxf(mx, sc[t][r]); }
#pragma unroll
      for (int off = 8; off > 0; off >>= 1) mx = fmaxf(mx, __shfl_xor(mx, off, 16));
      const float mnew  = fmaxf(mrow[r], mx);
      const float alpha = __expf(mrow[r] - mnew);
      mrow[r] = mnew;
      float rsum = 0.f;
#pragma unroll
      for (int t = 0; t < 4; ++t) {
        const float p = __expf(sc[t][r] - mnew);
        sc[t][r] = p;
        rsum += p;
      }
#pragma unroll
      for (int off = 8; off > 0; off >>= 1) rsum += __shfl_xor(rsum, off, 16);
      lrow[r] = lrow[r] * alpha + rsum;
#pragma unroll
      for (int t = 0; t < 4; ++t) Oacc[t][r] *= alpha;
    }

    // P (C layout) -> per-wave LDS -> A layout
#pragma unroll
    for (int t = 0; t < 4; ++t)
#pragma unroll
      for (int r = 0; r < 8; ++r)
        Pb[wid][rhalf + r][t * 16 + colg] = (_Float16)sc[t][r];
    const v16h Pa0 = load_a_frag(&Pb[wid][0][0], 72);   // keys 0..31
    const v16h Pa1 = load_a_frag(&Pb[wid][0][32], 72);  // keys 32..63

    // O(16x64) += P * V
#pragma unroll
    for (int t = 0; t < 4; ++t) {
      Oacc[t] = wmma_f16f32(Pa0, load_b_frag(&Vt[t * 16][0], 72), Oacc[t]);
      Oacc[t] = wmma_f16f32(Pa1, load_b_frag(&Vt[t * 16][32], 72), Oacc[t]);
    }
  }

  // normalize and store (b, q, h*64+d) as f16 for the out-projection GEMM
  const int q0 = qb * 128 + wid * 16;
#pragma unroll
  for (int t = 0; t < 4; ++t) {
#pragma unroll
    for (int r = 0; r < 8; ++r) {
      const int q = q0 + rhalf + r;
      const int d = t * 16 + colg;
      attnout[((size_t)b * SEQ + q) * INNER + h * DIM_HEAD + d] =
          (_Float16)(Oacc[t][r] / lrow[r]);
    }
  }
}

// ---------------------------------------------------------------------------
extern "C" void kernel_launch(void* const* d_in, const int* in_sizes, int n_in,
                              void* d_out, int out_size, void* d_ws, size_t ws_size,
                              hipStream_t stream) {
  (void)in_sizes; (void)n_in; (void)out_size; (void)ws_size;
  const float* x_in      = (const float*)d_in[0];
  const float* cb0       = (const float*)d_in[1];
  const float* cb1       = (const float*)d_in[2];
  const float* cb2       = (const float*)d_in[3];
  const float* cb3       = (const float*)d_in[4];
  const int*   codes_qkv = (const int*)d_in[5];
  const int*   codes_out = (const int*)d_in[6];
  const int*   codes_fc1 = (const int*)d_in[7];
  const int*   codes_fc2 = (const int*)d_in[8];
  const float* b_out     = (const float*)d_in[9];
  const float* b_fc1     = (const float*)d_in[10];
  const float* b_fc2     = (const float*)d_in[11];
  const float* ln1_w     = (const float*)d_in[12];
  const float* ln1_b     = (const float*)d_in[13];
  const float* ln2_w     = (const float*)d_in[14];
  const float* ln2_b     = (const float*)d_in[15];

  float* x = (float*)d_out;  // running residual stream (fp32), final output

  _Float16* ws = (_Float16*)d_ws;
  size_t off = 0;
  _Float16* wqkv = ws + off; off += (size_t)3 * INNER * DIM;   // 3072x1024
  _Float16* wout = ws + off; off += (size_t)DIM * INNER;       // 1024x1024
  _Float16* wfc1 = ws + off; off += (size_t)MLP * DIM;
  _Float16* wfc2 = ws + off; off += (size_t)DIM * MLP;
  _Float16* xn   = ws + off; off += (size_t)MTOT * DIM;        // LN output
  _Float16* qkvb = ws + off; off += (size_t)MTOT * 3 * INNER;  // qkv
  _Float16* attn = ws + off; off += (size_t)MTOT * INNER;      // attention out
  _Float16* hbuf = ws + off; off += (size_t)MTOT * MLP;        // gelu(fc1)

  hipMemcpyAsync(x, x_in, sizeof(float) * (size_t)MTOT * DIM,
                 hipMemcpyDeviceToDevice, stream);

  const dim3 blk(256);
  for (int i = 0; i < DEPTH; ++i) {
    build_weights<<<3 * INNER + DIM + MLP + DIM, blk, 0, stream>>>(
        cb0, cb1, cb2, cb3,
        codes_qkv + (size_t)i * 3 * INNER * HEADS,
        codes_out + (size_t)i * DIM * HEADS,
        codes_fc1 + (size_t)i * MLP * HEADS,
        codes_fc2 + (size_t)i * DIM * HEADS,
        wqkv, wout, wfc1, wfc2);

    layernorm_to_f16<<<MTOT, blk, 0, stream>>>(x, ln1_w + i * DIM, ln1_b + i * DIM, xn);

    gemm_wmma<<<dim3(3 * INNER / GBN, MTOT / GBM), blk, 0, stream>>>(
        xn, wqkv, nullptr, nullptr, nullptr, qkvb, MTOT, 3 * INNER, DIM, 0);

    attention_kernel<<<dim3(SEQ / 128, HEADS, BATCH), blk, 0, stream>>>(qkvb, attn);

    gemm_wmma<<<dim3(DIM / GBN, MTOT / GBM), blk, 0, stream>>>(
        attn, wout, b_out + (size_t)i * DIM, x, x, nullptr,
        MTOT, DIM, INNER, GF_BIAS | GF_RES | GF_F32);

    layernorm_to_f16<<<MTOT, blk, 0, stream>>>(x, ln2_w + i * DIM, ln2_b + i * DIM, xn);

    gemm_wmma<<<dim3(MLP / GBN, MTOT / GBM), blk, 0, stream>>>(
        xn, wfc1, b_fc1 + (size_t)i * MLP, nullptr, nullptr, hbuf,
        MTOT, MLP, DIM, GF_BIAS | GF_GELU);

    gemm_wmma<<<dim3(DIM / GBN, MTOT / GBM), blk, 0, stream>>>(
        hbuf, wfc2, b_fc2 + (size_t)i * DIM, x, x, nullptr,
        MTOT, DIM, MLP, GF_BIAS | GF_RES | GF_F32);
  }
}
